// Dynamics_5549097746481
// MI455X (gfx1250) — matrix operand order repaired and checked by reading, not compiled, not run on hardware
//
#include <hip/hip_runtime.h>

#define NBATCH 1024
#define NOBJ   32

typedef __attribute__((ext_vector_type(16))) _Float16 v16h;
typedef __attribute__((ext_vector_type(8)))  _Float16 h8;
typedef __attribute__((ext_vector_type(8)))  float    v8f;

union V16U { v16h v; h8 q[2]; };

struct Params {
  const float* s;
  const float* Wse; const float* bse;
  const float* sw0; const float* sb0;
  const float* sw1; const float* sb1;
  const float* rw0; const float* rb0;
  const float* rw1; const float* rb1;
  const float* rw2; const float* rb2;
  const float* aw0; const float* ab0;
  const float* aw1; const float* ab1;
  const float* aw2; const float* ab2;
  const float* fw0; const float* fb0;
  const float* fw1; const float* fb1;
  const float* fw2; const float* fb2;
  const float* ow0; const float* ob0;
  const float* ow1; const float* ob1;
  float* out;
};

// B-fragment (f16 32x16): lane = column n (&15); lanes 0-15 hold K 0..7,16..23,
// lanes 16-31 hold K 8..15,24..31. W row-major (fi, fo) f32 in global.
__device__ __forceinline__ v16h load_b_frag(const float* W, int fo, int kBase,
                                            int nBase, int lane) {
  int n  = nBase + (lane & 15);
  int k0 = kBase + ((lane >> 4) << 3);
  v16h b;
#pragma unroll
  for (int h = 0; h < 16; ++h) b[h] = (_Float16)W[(k0 + h + ((h >> 3) << 3)) * fo + n];
  return b;
}

// A-fragment (f16 16 x 32-K window): per lane two contiguous 8-half runs.
__device__ __forceinline__ v16h load_a_frag(const _Float16* kwin, int lane) {
  const int off = (lane >> 4) << 3;
  V16U A;
  A.q[0] = *(const h8*)(kwin + off);
  A.q[1] = *(const h8*)(kwin + 16 + off);
  return A.v;
}

// relu(a + c + dist*wd) A-fragment with packed f16 math.
__device__ __forceinline__ v16h build_pair_frag(const _Float16* aw, const _Float16* cw,
                                                const _Float16* wdw, _Float16 dij,
                                                int lane) {
  const int off = (lane >> 4) << 3;
  h8 z = {};
  h8 v0 = *(const h8*)(aw + off)      + *(const h8*)(cw + off)      + *(const h8*)(wdw + off)      * dij;
  h8 v1 = *(const h8*)(aw + 16 + off) + *(const h8*)(cw + 16 + off) + *(const h8*)(wdw + 16 + off) * dij;
  V16U A;
  A.q[0] = __builtin_elementwise_max(v0, z);
  A.q[1] = __builtin_elementwise_max(v1, z);
  return A.v;
}

__device__ __forceinline__ v8f wmma16(v16h a, v16h b, v8f c) {
  return __builtin_amdgcn_wmma_f32_16x16x32_f16(false, a, false, b,
                                                (short)0, c, false, false);
}

// Stage a weight matrix (count f32) from global into LDS.
__device__ __forceinline__ void stage_w(float* wstg, const float* Wg, int count, int tid) {
  for (int idx = tid; idx < count; idx += 256) wstg[idx] = Wg[idx];
}

// Scalar 32x32 @ 32x32: thread owns column c=tid&31 and 4 rows.
// ACT: 0 none, 1 relu, 2 tanh. resid (nullable) added after activation.
template <int ACT>
__device__ __forceinline__ void mm32(const float* lhs, const float* w,
                                     const float* bias, const float* resid,
                                     float* outb, int tid) {
  const int c = tid & 31, o0 = (tid >> 5) * 4;
  const float b = bias[c];
  float acc[4] = {b, b, b, b};
#pragma unroll 4
  for (int r = 0; r < 32; ++r) {
    float wv = w[r * 32 + c];
#pragma unroll
    for (int k = 0; k < 4; ++k) acc[k] += lhs[(o0 + k) * 32 + r] * wv;
  }
#pragma unroll
  for (int k = 0; k < 4; ++k) {
    float v = acc[k];
    if (ACT == 1) v = fmaxf(v, 0.f);
    if (ACT == 2) v = tanhf(v);
    if (resid) v += resid[(o0 + k) * 32 + c];
    outb[(o0 + k) * 32 + c] = v;
  }
}

__global__ __launch_bounds__(256) void dyn_net_kernel(Params P) {
  __shared__ float s2f[1024];    // s2
  __shared__ float distb[1024];  // dist; reused as aff3
  __shared__ float sdyn[1024];   // self_dyn; reused as aff2
  __shared__ float dynb[1024];   // dyn; reused as o1
  __shared__ float bufA[1024];   // h1; reused as aff1
  __shared__ float sbuf[512];    // raw s (32x16)
  __shared__ float att3w[8][32];
  __shared__ __align__(16) _Float16 s2h[1024];
  __shared__ __align__(16) _Float16 pairh[8192]; // [mat][32][64]: rel-a, rel-c, att-a, att-c
  __shared__ __align__(16) _Float16 wdrelh[64];
  __shared__ __align__(16) _Float16 wdatth[64];
  // 16KB scratch: rel2 f16 per-wave buffers during Phase B,
  // weight staging (f32, up to 2048) before/after Phase B.
  __shared__ __align__(16) unsigned char scratch16[16384];
  _Float16 (*rel2b)[1024] = (_Float16(*)[1024])scratch16;
  float* wstg = (float*)scratch16;

  const int tid  = threadIdx.x;
  const int lane = tid & 31;
  const int wave = tid >> 5;
  const int hi8  = (lane >> 4) << 3;
  const int n    = blockIdx.x;

  constexpr int REL_A = 0, REL_C = 2048, ATT_A = 4096, ATT_C = 6144;

  // ---------------- Phase A ----------------
  for (int idx = tid; idx < NOBJ * 16; idx += 256)
    sbuf[idx] = P.s[n * NOBJ * 16 + idx];
  stage_w(wstg, P.Wse, 16 * 32, tid);
  __syncthreads();

  // s2 = concat(s[:, :2], (s @ Wse + bse)[:, 2:]), f32 + f16 copies
  for (int idx = tid; idx < 1024; idx += 256) {
    int o = idx >> 5, c = idx & 31;
    float v;
    if (c < 2) {
      v = sbuf[o * 16 + c];
    } else {
      v = P.bse[c];
#pragma unroll
      for (int r = 0; r < 16; ++r) v += sbuf[o * 16 + r] * wstg[r * 32 + c];
    }
    s2f[idx] = v;
    s2h[idx] = (_Float16)v;
  }
  __syncthreads();

  // E3: stage sw0; pair_first a/c tables via WMMA; wd rows; dist
  stage_w(wstg, P.sw0, 1024, tid);
  {
    // wave -> (mat, jt); mat: 0 rel-a(+rb0), 1 rel-c, 2 att-a(+ab0), 3 att-c
    const int mat = wave >> 1;
    const int jt  = wave & 1;
    const float* Wb    = ((mat < 2) ? P.rw0 : P.aw0) + (mat & 1) * 32 * 64;
    const float* biasp = (mat < 2) ? P.rb0 : P.ab0;
    const float bscale = (mat & 1) ? 0.f : 1.f;
    _Float16* dst = pairh + mat * 2048;
    const int m = jt * 16 + (lane & 15);
    v16h As = load_a_frag(s2h + m * 32, lane);
#pragma unroll
    for (int nt = 0; nt < 4; ++nt) {
      v16h B = load_b_frag(Wb, 64, 0, nt * 16, lane);
      v8f acc = {};
      acc = wmma16(As, B, acc);
      int ch = nt * 16 + (lane & 15);
      float bch = biasp[ch] * bscale;
#pragma unroll
      for (int r = 0; r < 8; ++r) {
        int o = jt * 16 + r + hi8;
        dst[o * 64 + ch] = (_Float16)(acc[r] + bch);
      }
    }
  }
  if (tid < 64) {
    wdrelh[tid] = (_Float16)P.rw0[64 * 64 + tid];
    wdatth[tid] = (_Float16)P.aw0[64 * 64 + tid];
  }
  for (int idx = tid; idx < 1024; idx += 256) {
    int i = idx >> 5, j = idx & 31;
    float dx = s2f[i * 32 + 0] - s2f[j * 32 + 0];
    float dy = s2f[i * 32 + 1] - s2f[j * 32 + 1];
    distb[idx] = dx * dx + dy * dy;
  }
  __syncthreads();

  mm32<1>(s2f, wstg, P.sb0, nullptr, bufA, tid);     // h1 = relu(s2@sw0+sb0)
  __syncthreads();
  stage_w(wstg, P.sw1, 1024, tid);
  __syncthreads();
  mm32<0>(bufA, wstg, P.sb1, bufA, sdyn, tid);       // self_dyn = h1@sw1+sb1+h1

  // Per-wave B fragments (weights, f32 -> f16)
  v16h Ba1[2][2], Br1[2][2], Br2[2];
#pragma unroll
  for (int kt = 0; kt < 2; ++kt)
#pragma unroll
    for (int nt = 0; nt < 2; ++nt) {
      Ba1[kt][nt] = load_b_frag(P.aw1, 32, kt * 32, nt * 16, lane);
      Br1[kt][nt] = load_b_frag(P.rw1, 32, kt * 32, nt * 16, lane);
    }
#pragma unroll
  for (int nt = 0; nt < 2; ++nt) Br2[nt] = load_b_frag(P.rw2, 32, 0, nt * 16, lane);
  __syncthreads();

  // ---------------- Phase B: pairwise WMMA block ----------------
  const float ab2v = P.ab2[0];
  for (int it = 0; it < 4; ++it) {
    const int i = wave * 4 + it;

    // attention: att2 = relu(relu(att1) @ aw1 + ab1)
    v8f accA[2][2] = {};
#pragma unroll
    for (int jt = 0; jt < 2; ++jt) {
      const int m = jt * 16 + (lane & 15);
      const _Float16 dijh = (_Float16)distb[i * 32 + m];
#pragma unroll
      for (int kt = 0; kt < 2; ++kt) {
        v16h A = build_pair_frag(pairh + ATT_A + i * 64 + kt * 32,
                                 pairh + ATT_C + m * 64 + kt * 32,
                                 wdatth + kt * 32, dijh, lane);
#pragma unroll
        for (int nt = 0; nt < 2; ++nt)
          accA[jt][nt] = wmma16(A, Ba1[kt][nt], accA[jt][nt]);
      }
    }
    // att3[j] = exp(att2[j,:] . aw2 + ab2)
    float p[2][8] = {};
#pragma unroll
    for (int nt = 0; nt < 2; ++nt) {
      int ch = nt * 16 + (lane & 15);
      float b1 = P.ab1[ch], w2 = P.aw2[ch];
#pragma unroll
      for (int jt = 0; jt < 2; ++jt)
#pragma unroll
        for (int r = 0; r < 8; ++r)
          p[jt][r] += fmaxf(accA[jt][nt][r] + b1, 0.f) * w2;
    }
#pragma unroll
    for (int msk = 1; msk < 16; msk <<= 1)
#pragma unroll
      for (int jt = 0; jt < 2; ++jt)
#pragma unroll
        for (int r = 0; r < 8; ++r)
          p[jt][r] += __shfl_xor(p[jt][r], msk, 32);
    if ((lane & 15) == 0) {
#pragma unroll
      for (int jt = 0; jt < 2; ++jt)
#pragma unroll
        for (int r = 0; r < 8; ++r)
          att3w[wave][jt * 16 + r + hi8] = expf(p[jt][r] + ab2v);
    }
    __syncthreads();

    // rel: rel2 = relu(relu(rel1) @ rw1 + rb1)
    v8f accR[2][2] = {};
#pragma unroll
    for (int jt = 0; jt < 2; ++jt) {
      const int m = jt * 16 + (lane & 15);
      const _Float16 dijh = (_Float16)distb[i * 32 + m];
#pragma unroll
      for (int kt = 0; kt < 2; ++kt) {
        v16h A = build_pair_frag(pairh + REL_A + i * 64 + kt * 32,
                                 pairh + REL_C + m * 64 + kt * 32,
                                 wdrelh + kt * 32, dijh, lane);
#pragma unroll
        for (int nt = 0; nt < 2; ++nt)
          accR[jt][nt] = wmma16(A, Br1[kt][nt], accR[jt][nt]);
      }
    }
    // bias+relu in f32; stash f16 copy as next matmul's A operand
#pragma unroll
    for (int nt = 0; nt < 2; ++nt) {
      int ch = nt * 16 + (lane & 15);
      float b1 = P.rb1[ch];
#pragma unroll
      for (int jt = 0; jt < 2; ++jt)
#pragma unroll
        for (int r = 0; r < 8; ++r) {
          float v = fmaxf(accR[jt][nt][r] + b1, 0.f);
          accR[jt][nt][r] = v;
          int j = jt * 16 + r + hi8;
          rel2b[wave][j * 32 + ch] = (_Float16)v;
        }
    }
    __syncthreads();

    // rel_f = rel2 @ rw2 + rb2 + rel2 (C = f32 rel2 residual)
    v8f accF[2][2];
#pragma unroll
    for (int jt = 0; jt < 2; ++jt) {
      const int m = jt * 16 + (lane & 15);
      v16h A2 = load_a_frag(rel2b[wave] + m * 32, lane);
#pragma unroll
      for (int nt = 0; nt < 2; ++nt)
        accF[jt][nt] = wmma16(A2, Br2[nt], accR[jt][nt]);
    }

    // rel_dyn[i] = sum_{j != i} att3[j] * rel_f[i,j,:]; dyn = self_dyn + rel_dyn
#pragma unroll
    for (int nt = 0; nt < 2; ++nt) {
      int ch = nt * 16 + (lane & 15);
      float b2 = P.rb2[ch];
      float tot = 0.f;
#pragma unroll
      for (int jt = 0; jt < 2; ++jt)
#pragma unroll
        for (int r = 0; r < 8; ++r) {
          int j = jt * 16 + r + hi8;
          float w = (j == i) ? 0.f : att3w[wave][j];
          tot += (accF[jt][nt][r] + b2) * w;
        }
      tot += __shfl_xor(tot, 16, 32);
      if (lane < 16) dynb[i * 32 + ch] = sdyn[i * 32 + ch] + tot;
    }
  }
  __syncthreads();

  // ---------------- Phase C: output chain (scalar f32, LDS-staged weights) ----
  stage_w(wstg, P.fw0, 1024, tid);
  __syncthreads();
  mm32<2>(dynb, wstg, P.fb0, nullptr, bufA, tid);    // aff1 = tanh(dyn@fw0+fb0)
  __syncthreads();
  stage_w(wstg, P.fw1, 1024, tid);
  __syncthreads();
  mm32<2>(bufA, wstg, P.fb1, bufA, sdyn, tid);       // aff2 = tanh(...)+aff1
  __syncthreads();
  stage_w(wstg, P.fw2, 1024, tid);
  __syncthreads();
  mm32<0>(sdyn, wstg, P.fb2, nullptr, distb, tid);   // aff3
  __syncthreads();
  stage_w(wstg, P.ow0, 2048, tid);
  __syncthreads();
  {  // o1 = tanh([aff3, s2] @ ow0 + ob0)
    const int c = tid & 31, o0 = (tid >> 5) * 4;
    const float b = P.ob0[c];
    float acc[4] = {b, b, b, b};
#pragma unroll 4
    for (int r = 0; r < 32; ++r) {
      float w1 = wstg[r * 32 + c];
      float w2 = wstg[(32 + r) * 32 + c];
#pragma unroll
      for (int k = 0; k < 4; ++k)
        acc[k] += distb[(o0 + k) * 32 + r] * w1 + s2f[(o0 + k) * 32 + r] * w2;
    }
#pragma unroll
    for (int k = 0; k < 4; ++k) dynb[(o0 + k) * 32 + c] = tanhf(acc[k]);
  }
  __syncthreads();
  stage_w(wstg, P.ow1, 1024, tid);
  __syncthreads();
  {  // result = o1 @ ow1 + ob1 + o1
    const int c = tid & 31, o0 = (tid >> 5) * 4;
    const float b = P.ob1[c];
    float acc[4] = {b, b, b, b};
#pragma unroll 4
    for (int r = 0; r < 32; ++r) {
      float wv = wstg[r * 32 + c];
#pragma unroll
      for (int k = 0; k < 4; ++k) acc[k] += dynb[(o0 + k) * 32 + r] * wv;
    }
#pragma unroll
    for (int k = 0; k < 4; ++k)
      P.out[n * 1024 + (o0 + k) * 32 + c] = acc[k] + dynb[(o0 + k) * 32 + c];
  }
}

extern "C" void kernel_launch(void* const* d_in, const int* in_sizes, int n_in,
                              void* d_out, int out_size, void* d_ws, size_t ws_size,
                              hipStream_t stream) {
  (void)in_sizes; (void)n_in; (void)out_size; (void)d_ws; (void)ws_size;
  Params P;
  P.s   = (const float*)d_in[0];
  P.Wse = (const float*)d_in[1];  P.bse = (const float*)d_in[2];
  P.sw0 = (const float*)d_in[3];  P.sb0 = (const float*)d_in[4];
  P.sw1 = (const float*)d_in[5];  P.sb1 = (const float*)d_in[6];
  P.rw0 = (const float*)d_in[7];  P.rb0 = (const float*)d_in[8];
  P.rw1 = (const float*)d_in[9];  P.rb1 = (const float*)d_in[10];
  P.rw2 = (const float*)d_in[11]; P.rb2 = (const float*)d_in[12];
  P.aw0 = (const float*)d_in[13]; P.ab0 = (const float*)d_in[14];
  P.aw1 = (const float*)d_in[15]; P.ab1 = (const float*)d_in[16];
  P.aw2 = (const float*)d_in[17]; P.ab2 = (const float*)d_in[18];
  P.fw0 = (const float*)d_in[19]; P.fb0 = (const float*)d_in[20];
  P.fw1 = (const float*)d_in[21]; P.fb1 = (const float*)d_in[22];
  P.fw2 = (const float*)d_in[23]; P.fb2 = (const float*)d_in[24];
  P.ow0 = (const float*)d_in[25]; P.ob0 = (const float*)d_in[26];
  P.ow1 = (const float*)d_in[27]; P.ob1 = (const float*)d_in[28];
  // d_in[29] = core_idx (unused)
  P.out = (float*)d_out;
  dyn_net_kernel<<<NBATCH, 256, 0, stream>>>(P);
}
